// ShapeNetModelSeg_62148176773701
// MI455X (gfx1250) — matrix operand order repaired and checked by reading, not compiled, not run on hardware
//
#include <hip/hip_runtime.h>
#include <hip/hip_bf16.h>

// ---------------- problem constants (match reference) ----------------
#define BB   8
#define CCH  128
#define NPTS 2048
#define KNN  32
#define FFC  512   // 4*C
#define BNEPS 1e-5f

typedef __attribute__((ext_vector_type(16))) __bf16 v16bf;
typedef __attribute__((ext_vector_type(8)))  float  v8f;

// =====================================================================
// 1) KNN: one thread per point, LDS-tiled scan, insertion list of 32.
//    grid = B*(N/256), 256 threads
// =====================================================================
__global__ void knn_kernel(const float* __restrict__ coord, int* __restrict__ idx_out) {
    const int b     = blockIdx.x / (NPTS / 256);
    const int tile0 = (blockIdx.x % (NPTS / 256)) * 256;
    const int n     = tile0 + threadIdx.x;
    const float* cx = coord + (size_t)b * 3 * NPTS;
    const float px = cx[n], py = cx[NPTS + n], pz = cx[2 * NPTS + n];

    float bestd[KNN];
    int   besti[KNN];
    for (int j = 0; j < KNN; ++j) { bestd[j] = 3.4e38f; besti[j] = 0; }

    __shared__ float sx[256], sy[256], sz[256];
    for (int m0 = 0; m0 < NPTS; m0 += 256) {
        __syncthreads();
        sx[threadIdx.x] = cx[m0 + threadIdx.x];
        sy[threadIdx.x] = cx[NPTS + m0 + threadIdx.x];
        sz[threadIdx.x] = cx[2 * NPTS + m0 + threadIdx.x];
        __syncthreads();
        for (int mm = 0; mm < 256; ++mm) {
            const float dx = px - sx[mm], dy = py - sy[mm], dz = pz - sz[mm];
            const float d2 = dx * dx + dy * dy + dz * dz;
            if (d2 < bestd[KNN - 1]) {
                int j = KNN - 1;
                while (j > 0 && bestd[j - 1] > d2) {
                    bestd[j] = bestd[j - 1]; besti[j] = besti[j - 1]; --j;
                }
                bestd[j] = d2; besti[j] = m0 + mm;
            }
        }
    }
    int* out = idx_out + ((size_t)b * NPTS + n) * KNN;
    for (int j = 0; j < KNN; ++j) out[j] = besti[j];
}

// ---- bf16 packing helpers (emit v_cvt_pk_bf16_f32 patterns) ----
__device__ __forceinline__ unsigned pack_bf16x2(float lo, float hi) {
    union { __bf16 h[2]; unsigned u; } pk;
    pk.h[0] = (__bf16)lo; pk.h[1] = (__bf16)hi;
    return pk.u;
}
__device__ __forceinline__ unsigned long long pack_bf16x4(float a, float b, float c, float d) {
    union { __bf16 h[4]; unsigned long long u; } pk;
    pk.h[0] = (__bf16)a; pk.h[1] = (__bf16)b; pk.h[2] = (__bf16)c; pk.h[3] = (__bf16)d;
    return pk.u;
}

// =====================================================================
// 2) Per-batch channel GEMM, bf16 WMMA with f32 accumulate.
//    OUT[b,o,n] (or [b,n,o]) = act( sum_c W[o,c] * X[b,c,n] ) (+ ADD)
//    block = 128 thr (4 waves); wave owns 16(o) x 16(n), block 16 x 64.
//    LDS tiles are stored in WMMA *fragment order* so each lane loads
//    its whole fragment as 32 contiguous, 32B-aligned bytes.
// =====================================================================
template <int CIN, int ACT, bool NC, bool HASADD>
__global__ void gemm_wmma_kernel(const float* __restrict__ W,
                                 const float* __restrict__ X,
                                 const float* __restrict__ ADD,   // (B,O,N) or null
                                 float* __restrict__ OUT,
                                 int O) {
    const int Nn   = NPTS;
    const int b    = blockIdx.z;
    const int o0   = blockIdx.y * 16;
    const int n0   = blockIdx.x * 64;
    const int wave = threadIdx.x >> 5;
    const int lane = threadIdx.x & 31;
    const int wn0  = wave * 16;
    const float* Xb = X + (size_t)b * CIN * Nn;

    // fragment-order LDS tiles
    __shared__ __attribute__((aligned(32))) __bf16 sAf[32][16];  // [lane][elem]
    __shared__ __attribute__((aligned(32))) __bf16 sBf[64][32];  // [col][k] (col-major X tile)

    // per-thread constant staging coordinates
    // A: thread t stages 4 elems of consumer-lane (t>>2) at elems (t&3)*4..+3
    const int a_lane = threadIdx.x >> 2;
    const int a_e0   = (threadIdx.x & 3) * 4;
    const int a_row  = a_lane & 15;
    const int a_kb0  = (a_lane < 16) ? 0 : 16;
    const int a_k    = a_kb0 + a_e0;

    // fragment read coordinates
    const int f_col = wn0 + (lane & 15);
    const int f_kb  = (lane < 16) ? 0 : 16;

    v8f acc = {};
    for (int c0 = 0; c0 < CIN; c0 += 32) {
        __syncthreads();
        // ---- stage A tile (16x32 weights) into fragment order ----
        {
            const float4 w4 = *(const float4*)(W + (size_t)(o0 + a_row) * CIN + c0 + a_k);
            *(unsigned long long*)&sAf[a_lane][a_e0] = pack_bf16x4(w4.x, w4.y, w4.z, w4.w);
        }
        // ---- stage B tile (32x64 acts) transposed: 2(k) x 4(n) micro-tiles ----
        #pragma unroll
        for (int mtb = 0; mtb < 2; ++mtb) {
            const int mt = mtb * 128 + threadIdx.x;
            const int r  = (mt >> 4) * 2;       // k: 0,2,...,30
            const int c  = (mt & 15) * 4;       // n: 0,4,...,60
            const float4 x0 = *(const float4*)(Xb + (size_t)(c0 + r)     * Nn + (n0 + c));
            const float4 x1 = *(const float4*)(Xb + (size_t)(c0 + r + 1) * Nn + (n0 + c));
            *(unsigned*)&sBf[c + 0][r] = pack_bf16x2(x0.x, x1.x);
            *(unsigned*)&sBf[c + 1][r] = pack_bf16x2(x0.y, x1.y);
            *(unsigned*)&sBf[c + 2][r] = pack_bf16x2(x0.z, x1.z);
            *(unsigned*)&sBf[c + 3][r] = pack_bf16x2(x0.w, x1.w);
        }
        if (c0 + 32 < CIN)
            __builtin_prefetch(Xb + (size_t)(c0 + 32 + (threadIdx.x >> 2)) * Nn + n0, 0, 1);
        __syncthreads();

        // ---- wide fragment loads (2x ds_load_b128 each) ----
        const v16bf afrag = *(const v16bf*)&sAf[lane][0];
        const v16bf bfrag = *(const v16bf*)&sBf[f_col][f_kb];

        acc = __builtin_amdgcn_wmma_f32_16x16x32_bf16(
                  false, afrag, false, bfrag, (short)0, acc, false, false);
    }

    // D 16x16 f32: VGPR r -> M = r + 8*(lane>=16), N = lane&15
    #pragma unroll
    for (int r = 0; r < 8; ++r) {
        const int orow = o0 + r + ((lane < 16) ? 0 : 8);
        const int ncol = n0 + wn0 + (lane & 15);
        float vv = acc[r];
        if (ACT == 1) vv = (vv >= 0.f) ? vv : 0.2f * vv;
        if (HASADD) vv += ADD[((size_t)b * O + orow) * Nn + ncol];
        if (NC)
            OUT[((size_t)b * Nn + ncol) * O + orow] = vv;   // (B,N,O)
        else
            OUT[((size_t)b * O + orow) * Nn + ncol] = vv;   // (B,O,N)
    }
}

// =====================================================================
// 3) Local attention, one wave per point; lane owns 4 channels.
//    k_diff = Kp[idx]-Kp[n], v_diff = Vp[idx]-Vp[n] (linearity).
// =====================================================================
__global__ void attn_kernel(const float* __restrict__ pcd,
                            const float* __restrict__ Qp,   // (B,N,C)
                            const float* __restrict__ Kp,   // (B,N,C)
                            const float* __restrict__ Vp,   // (B,N,C)
                            const int*   __restrict__ idx,  // (B,N,K)
                            float* __restrict__ xpre) {     // (B,C,N)
    const int wave = threadIdx.x >> 5;
    const int lane = threadIdx.x & 31;
    const int p = blockIdx.x * 8 + wave;
    const int b = p / NPTS, n = p % NPTS;
    const size_t rowbase = ((size_t)b * NPTS + n) * CCH;
    const int c0 = lane * 4;

    float qv[4], kc[4];
    #pragma unroll
    for (int i = 0; i < 4; ++i) qv[i] = Qp[rowbase + c0 + i];
    #pragma unroll
    for (int i = 0; i < 4; ++i) kc[i] = Kp[rowbase + c0 + i];
    const int myidx = idx[((size_t)b * NPTS + n) * KNN + lane];

    float e[KNN];
    #pragma unroll
    for (int j = 0; j < KNN; ++j) {
        const int m = __shfl(myidx, j, 32);
        const float* kr = Kp + ((size_t)b * NPTS + m) * CCH + c0;
        float s = 0.f;
        #pragma unroll
        for (int i = 0; i < 4; ++i) s += qv[i] * (kr[i] - kc[i]);
        s += __shfl_xor(s, 1, 32);
        s += __shfl_xor(s, 2, 32);
        e[j] = s * 0.25f;   // 1/sqrt(d), d=16
    }
    float mx = -3.4e38f;
    #pragma unroll
    for (int j = 0; j < KNN; ++j) mx = fmaxf(mx, e[j]);
    float sum = 0.f;
    #pragma unroll
    for (int j = 0; j < KNN; ++j) { e[j] = __expf(e[j] - mx); sum += e[j]; }
    const float inv = 1.f / sum;

    float acc[4] = {0.f, 0.f, 0.f, 0.f}, vc[4];
    #pragma unroll
    for (int i = 0; i < 4; ++i) vc[i] = Vp[rowbase + c0 + i];
    #pragma unroll
    for (int j = 0; j < KNN; ++j) {
        const int m = __shfl(myidx, j, 32);
        const float* vr = Vp + ((size_t)b * NPTS + m) * CCH + c0;
        const float a = e[j] * inv;
        #pragma unroll
        for (int i = 0; i < 4; ++i) acc[i] += a * (vr[i] - vc[i]);
    }
    #pragma unroll
    for (int i = 0; i < 4; ++i) {
        const size_t o = ((size_t)b * CCH + (c0 + i)) * NPTS + n;
        xpre[o] = pcd[o] + acc[i];
    }
}

// =====================================================================
// 4) BatchNorm stats per channel over (B,N): grid = C, block = 256
// =====================================================================
__global__ void bnstats_kernel(const float* __restrict__ x,
                               float* __restrict__ mean, float* __restrict__ rstd) {
    const int c = blockIdx.x, t = threadIdx.x;
    float s = 0.f, s2 = 0.f;
    for (int i = t; i < BB * NPTS; i += 256) {
        const int b = i / NPTS, n = i % NPTS;
        const float v = x[((size_t)b * CCH + c) * NPTS + n];
        s += v; s2 += v * v;
    }
    __shared__ float rs[256], rs2[256];
    rs[t] = s; rs2[t] = s2;
    __syncthreads();
    for (int k = 128; k > 0; k >>= 1) {
        if (t < k) { rs[t] += rs[t + k]; rs2[t] += rs2[t + k]; }
        __syncthreads();
    }
    if (t == 0) {
        const float m = rs[0] * (1.f / (BB * NPTS));
        const float var = rs2[0] * (1.f / (BB * NPTS)) - m * m;
        mean[c] = m;
        rstd[c] = rsqrtf(var + BNEPS);
    }
}

// =====================================================================
// 5) BatchNorm apply (elementwise)
// =====================================================================
__global__ void bnapply_kernel(const float* __restrict__ x,
                               const float* __restrict__ mean,
                               const float* __restrict__ rstd,
                               const float* __restrict__ g,
                               const float* __restrict__ beta,
                               float* __restrict__ out) {
    const size_t i = (size_t)blockIdx.x * 256 + threadIdx.x;
    const int c = (int)((i / NPTS) % CCH);
    out[i] = (x[i] - mean[c]) * rstd[c] * g[c] + beta[c];
}

// =====================================================================
// launcher
// =====================================================================
extern "C" void kernel_launch(void* const* d_in, const int* in_sizes, int n_in,
                              void* d_out, int out_size, void* d_ws, size_t ws_size,
                              hipStream_t stream) {
    const float* pcd   = (const float*)d_in[0];
    const float* coord = (const float*)d_in[1];
    const float* wq    = (const float*)d_in[2];
    const float* wk    = (const float*)d_in[3];
    const float* wv    = (const float*)d_in[4];
    const float* ffw1  = (const float*)d_in[5];
    const float* ffw2  = (const float*)d_in[6];
    const float* bn1g  = (const float*)d_in[7];
    const float* bn1b  = (const float*)d_in[8];
    const float* bn2g  = (const float*)d_in[9];
    const float* bn2b  = (const float*)d_in[10];
    float* out = (float*)d_out;

    char* base = (char*)d_ws;
    size_t off = 0;
    auto carve = [&](size_t bytes) -> char* {
        char* p = base + off;
        off = (off + bytes + 255) & ~(size_t)255;
        return p;
    };
    int*   idx  = (int*)  carve((size_t)BB * NPTS * KNN * 4);   //  2 MiB
    float* Qp   = (float*)carve((size_t)BB * NPTS * CCH * 4);   //  8 MiB
    float* Kp   = (float*)carve((size_t)BB * NPTS * CCH * 4);
    float* Vp   = (float*)carve((size_t)BB * NPTS * CCH * 4);
    float* xpre = (float*)carve((size_t)BB * CCH * NPTS * 4);
    float* x    = (float*)carve((size_t)BB * CCH * NPTS * 4);
    float* h1   = (float*)carve((size_t)BB * FFC * NPTS * 4);   // 32 MiB
    float* ypre = (float*)carve((size_t)BB * CCH * NPTS * 4);
    float* m1   = (float*)carve(CCH * 4);
    float* r1   = (float*)carve(CCH * 4);
    float* m2   = (float*)carve(CCH * 4);
    float* r2   = (float*)carve(CCH * 4);
    (void)ws_size; (void)n_in; (void)in_sizes; (void)out_size;

    // 1) KNN
    knn_kernel<<<dim3(BB * (NPTS / 256)), dim3(256), 0, stream>>>(coord, idx);

    // 2) Q/K/V projections -> (B,N,C) for contiguous neighbor-row gathers
    {
        dim3 g(NPTS / 64, CCH / 16, BB), bdim(128);
        gemm_wmma_kernel<CCH, 0, true, false><<<g, bdim, 0, stream>>>(wq, pcd, nullptr, Qp, CCH);
        gemm_wmma_kernel<CCH, 0, true, false><<<g, bdim, 0, stream>>>(wk, pcd, nullptr, Kp, CCH);
        gemm_wmma_kernel<CCH, 0, true, false><<<g, bdim, 0, stream>>>(wv, pcd, nullptr, Vp, CCH);
    }

    // 3) attention + residual -> xpre (B,C,N)
    attn_kernel<<<dim3(BB * NPTS / 8), dim3(256), 0, stream>>>(pcd, Qp, Kp, Vp, idx, xpre);

    // 4) BN1
    bnstats_kernel<<<dim3(CCH), dim3(256), 0, stream>>>(xpre, m1, r1);
    bnapply_kernel<<<dim3((unsigned)((size_t)BB * CCH * NPTS / 256)), dim3(256), 0, stream>>>(
        xpre, m1, r1, bn1g, bn1b, x);

    // 5) feed-forward: h1 = lrelu(ff_w1 @ x), ypre = ff_w2 @ h1 + x
    {
        dim3 g1(NPTS / 64, FFC / 16, BB), bdim(128);
        gemm_wmma_kernel<CCH, 1, false, false><<<g1, bdim, 0, stream>>>(ffw1, x, nullptr, h1, FFC);
        dim3 g2(NPTS / 64, CCH / 16, BB);
        gemm_wmma_kernel<FFC, 0, false, true><<<g2, bdim, 0, stream>>>(ffw2, h1, x, ypre, CCH);
    }

    // 6) BN2 -> final output
    bnstats_kernel<<<dim3(CCH), dim3(256), 0, stream>>>(ypre, m2, r2);
    bnapply_kernel<<<dim3((unsigned)((size_t)BB * CCH * NPTS / 256)), dim3(256), 0, stream>>>(
        ypre, m2, r2, bn2g, bn2b, out);
}
